// LSTM_6305011990844
// MI455X (gfx1250) — matrix operand order repaired and checked by reading, not compiled, not run on hardware
//
#include <hip/hip_runtime.h>
#include <hip/hip_bf16.h>

// ---------------- problem constants ----------------
#define Bsz  32
#define Tlen 1024
#define Dd   512
#define Hh   1024
#define NWG  64          // persistent workgroups, 16 H-columns each

// ---------------- workspace layout (bytes) ----------------
static constexpr size_t OFF_BAR = 0;                    // 2 x u32 barrier
static constexpr size_t OFF_H   = 1024;                 // h double buffer: 2*B*H bf16 = 128KB
static constexpr size_t OFF_XBF = (size_t)1 << 20;      // x bf16 [T][B][D] = 32MB
static constexpr size_t OFF_WXT = (size_t)36 << 20;     // WxT bf16 [4][H][D] = 4MB
static constexpr size_t OFF_WHT = (size_t)42 << 20;     // WhT bf16 [4][H][H] = 8MB

typedef __attribute__((ext_vector_type(16))) __bf16 v16bf;
typedef __attribute__((ext_vector_type(8)))  float  v8f;

// A-fragment (16x32 bf16, M=batch rows): lane(0-15)=row, holds K [p,p+8) and [p+16,p+24)
__device__ __forceinline__ v16bf loadA(const __bf16* p) {
  union { uint4 q[2]; v16bf v; } u;
  u.q[0] = *reinterpret_cast<const uint4*>(p);
  u.q[1] = *reinterpret_cast<const uint4*>(p + 16);
  return u.v;
}
// B-fragment (32x16 bf16, N=cols): lane(0-15)=col, holds K [p,p+8) and [p+16,p+24)
__device__ __forceinline__ v16bf loadB(const __bf16* p) {
  union { uint4 q[2]; v16bf v; } u;
  u.q[0] = *reinterpret_cast<const uint4*>(p);
  u.q[1] = *reinterpret_cast<const uint4*>(p + 16);
  return u.v;
}

__device__ __forceinline__ float sigmoidf_(float x) {
  return 1.0f / (1.0f + __expf(-x));
}

// device-wide generation barrier (counter + gen live in L2-resident workspace)
__device__ __forceinline__ void grid_barrier(unsigned* bar, unsigned nwg, unsigned target) {
  __syncthreads();
  if (threadIdx.x == 0) {
    __threadfence();
    unsigned prev = __hip_atomic_fetch_add(&bar[0], 1u, __ATOMIC_ACQ_REL, __HIP_MEMORY_SCOPE_AGENT);
    if (prev == nwg - 1u) {
      __hip_atomic_store(&bar[0], 0u, __ATOMIC_RELAXED, __HIP_MEMORY_SCOPE_AGENT);
      __hip_atomic_fetch_add(&bar[1], 1u, __ATOMIC_RELEASE, __HIP_MEMORY_SCOPE_AGENT);
    }
    while (__hip_atomic_load(&bar[1], __ATOMIC_ACQUIRE, __HIP_MEMORY_SCOPE_AGENT) < target) {
      __builtin_amdgcn_s_sleep(8);
    }
    __threadfence();
  }
  __syncthreads();
}

// ---------------- setup kernels ----------------
__global__ void k_zero(unsigned* __restrict__ bar, __bf16* __restrict__ hb, int n) {
  int i = blockIdx.x * blockDim.x + threadIdx.x;
  if (i < n) hb[i] = (__bf16)0.0f;
  if (i == 0) { bar[0] = 0u; bar[1] = 0u; }
}

// x [B,T,D] f32 -> xbf [T,B,D] bf16 (coalesced along D on both sides)
__global__ void k_cvt_x(const float* __restrict__ x, __bf16* __restrict__ xbf) {
  size_t i = (size_t)blockIdx.x * 256 + threadIdx.x;   // < T*B*D
  int d = (int)(i & (Dd - 1));
  size_t rem = i >> 9;                                  // /D
  int b = (int)(rem & (Bsz - 1));
  int t = (int)(rem >> 5);                              // /B
  xbf[i] = (__bf16)x[(((size_t)b * Tlen + t) << 9) + d];
}

// W [K,H] f32 row-major -> WT [H][K] bf16 (column-major per gate)
__global__ void k_cvt_w(const float* __restrict__ w, __bf16* __restrict__ wt, int K) {
  size_t i = (size_t)blockIdx.x * 256 + threadIdx.x;   // < H*K
  int k   = (int)(i % (size_t)K);
  int col = (int)(i / (size_t)K);
  wt[i] = (__bf16)w[(size_t)k * Hh + col];
}

// ---------------- persistent LSTM kernel ----------------
__global__ __launch_bounds__(256) void k_lstm(
    const __bf16* __restrict__ xbf,   // [T][B][D]
    const __bf16* __restrict__ whT,   // [4][H cols][H k]
    const __bf16* __restrict__ wxT,   // [4][H cols][D k]
    const float* __restrict__ bfb, const float* __restrict__ bib,
    const float* __restrict__ bob, const float* __restrict__ bcb,
    __bf16* __restrict__ hbuf,        // [2][B][H] bf16 (double buffer)
    unsigned* __restrict__ bar,
    float* __restrict__ out)          // h_seq [B,T,H] ++ hT [B,H] ++ cT [B,H]
{
  __shared__ float stage[4][32][16];  // [gate][batch row][col] f32 gate pre-activations

  const int tid    = threadIdx.x;
  const int lane   = tid & 31;
  const int l15    = lane & 15;
  const int laneHi = (lane >> 4) & 1;
  const int wave   = tid >> 5;        // 0..7
  const int g      = wave >> 1;       // gate 0..3 (f,i,o,c)
  const int mt     = wave & 1;        // M-tile (batch rows 0-15 / 16-31)
  const int colBase = blockIdx.x * 16;

  // WMMA-phase addressing
  const int arow = mt * 16 + l15;                              // batch row for A fragment
  const int offA = laneHi << 3;                                // +8 elements for upper half-wave
  const int offB = laneHi << 3;                                // +8 elements for upper half-wave
  const __bf16* wxp = wxT + ((size_t)g * Hh + colBase + l15) * Dd + offB;
  const __bf16* whp = whT + ((size_t)g * Hh + colBase + l15) * Hh + offB;

  // elementwise-phase mapping: thread -> (rows erow, erow+16) x column ecol
  const int ecol  = tid & 15;
  const int erow  = tid >> 4;          // 0..15
  const int ecolg = colBase + ecol;
  const float bfv = bfb[ecolg], biv = bib[ecolg], bov = bob[ecolg], bcv = bcb[ecolg];
  float cst[2] = {0.0f, 0.0f};         // cell state, persistent across all T steps

  for (int t = 0; t < Tlen; ++t) {
    v8f acc0 = {}; v8f acc1 = {};

    // ---- input projection part: A = x_t (16 x D), B = WxT slice ----
    {
      const __bf16* ap = xbf + ((size_t)t * Bsz + arow) * Dd + offA;
      for (int k = 0; k < Dd; k += 64) {
        acc0 = __builtin_amdgcn_wmma_f32_16x16x32_bf16(
                 false, loadA(ap + k),      false, loadB(wxp + k),
                 (short)0, acc0, false, false);
        acc1 = __builtin_amdgcn_wmma_f32_16x16x32_bf16(
                 false, loadA(ap + k + 32), false, loadB(wxp + k + 32),
                 (short)0, acc1, false, false);
      }
    }
    // ---- recurrent part: A = h_t (16 x H), B = WhT slice ----
    {
      const __bf16* hp = hbuf + ((size_t)(t & 1) * Bsz + arow) * Hh + offA;
      for (int k = 0; k < Hh; k += 64) {
        acc0 = __builtin_amdgcn_wmma_f32_16x16x32_bf16(
                 false, loadA(hp + k),      false, loadB(whp + k),
                 (short)0, acc0, false, false);
        acc1 = __builtin_amdgcn_wmma_f32_16x16x32_bf16(
                 false, loadA(hp + k + 32), false, loadB(whp + k + 32),
                 (short)0, acc1, false, false);
      }
    }

    // ---- stage 16x16 f32 tile: VGPR i holds (row i | i+8), col = lane%16 ----
    {
      v8f s = acc0 + acc1;
      float* st = &stage[g][mt * 16 + laneHi * 8][l15];
      #pragma unroll
      for (int i = 0; i < 8; ++i) st[i * 16] = s[i];
    }
    __syncthreads();

    // ---- elementwise LSTM cell update (2 cells per thread) ----
    __bf16* hn = hbuf + (size_t)((t + 1) & 1) * Bsz * Hh;
    #pragma unroll
    for (int half = 0; half < 2; ++half) {
      const int r = erow + half * 16;   // batch row
      float f  = sigmoidf_(stage[0][r][ecol] + bfv);
      float ii = sigmoidf_(stage[1][r][ecol] + biv);
      float oo = sigmoidf_(stage[2][r][ecol] + bov);
      float gg = tanhf    (stage[3][r][ecol] + bcv);
      float c  = f * cst[half] + ii * gg;
      cst[half] = c;
      float hh = tanhf(c) * oo;
      out[((size_t)r * Tlen + t) * Hh + ecolg] = hh;         // h_seq [B,T,H]
      hn[(size_t)r * Hh + ecolg] = (__bf16)hh;               // next-step h (bf16)
      if (t == Tlen - 1) {
        const size_t tail = (size_t)Bsz * Tlen * Hh;
        out[tail + (size_t)r * Hh + ecolg] = hh;                         // hT
        out[tail + (size_t)Bsz * Hh + (size_t)r * Hh + ecolg] = c;       // cT
      }
    }

    if (t + 1 < Tlen) grid_barrier(bar, NWG, (unsigned)(t + 1));
    else              __syncthreads();
  }
}

// ---------------- host launcher ----------------
extern "C" void kernel_launch(void* const* d_in, const int* in_sizes, int n_in,
                              void* d_out, int out_size, void* d_ws, size_t ws_size,
                              hipStream_t stream) {
  (void)in_sizes; (void)n_in; (void)out_size; (void)ws_size;
  const float* x   = (const float*)d_in[0];
  const float* wfx = (const float*)d_in[1];
  const float* wfh = (const float*)d_in[2];
  const float* bf  = (const float*)d_in[3];
  const float* wix = (const float*)d_in[4];
  const float* wih = (const float*)d_in[5];
  const float* bi  = (const float*)d_in[6];
  const float* wox = (const float*)d_in[7];
  const float* woh = (const float*)d_in[8];
  const float* bo  = (const float*)d_in[9];
  const float* wcx = (const float*)d_in[10];
  const float* wch = (const float*)d_in[11];
  const float* bc  = (const float*)d_in[12];

  char* ws = (char*)d_ws;
  unsigned* bar = (unsigned*)(ws + OFF_BAR);
  __bf16* hbuf  = (__bf16*)(ws + OFF_H);
  __bf16* xbf   = (__bf16*)(ws + OFF_XBF);
  __bf16* wxT   = (__bf16*)(ws + OFF_WXT);
  __bf16* whT   = (__bf16*)(ws + OFF_WHT);
  float*  out   = (float*)d_out;

  // 1) zero barrier + initial hidden state
  k_zero<<<(2 * Bsz * Hh) / 256, 256, 0, stream>>>(bar, hbuf, 2 * Bsz * Hh);

  // 2) x -> bf16 [T,B,D]
  k_cvt_x<<<(Tlen * Bsz * Dd) / 256, 256, 0, stream>>>(x, xbf);

  // 3) weights -> bf16, column-major [gate][col][k]
  const size_t wxg = (size_t)Hh * Dd;
  const size_t whg = (size_t)Hh * Hh;
  k_cvt_w<<<(Hh * Dd) / 256, 256, 0, stream>>>(wfx, wxT + 0 * wxg, Dd);
  k_cvt_w<<<(Hh * Dd) / 256, 256, 0, stream>>>(wix, wxT + 1 * wxg, Dd);
  k_cvt_w<<<(Hh * Dd) / 256, 256, 0, stream>>>(wox, wxT + 2 * wxg, Dd);
  k_cvt_w<<<(Hh * Dd) / 256, 256, 0, stream>>>(wcx, wxT + 3 * wxg, Dd);
  k_cvt_w<<<(Hh * Hh) / 256, 256, 0, stream>>>(wfh, whT + 0 * whg, Hh);
  k_cvt_w<<<(Hh * Hh) / 256, 256, 0, stream>>>(wih, whT + 1 * whg, Hh);
  k_cvt_w<<<(Hh * Hh) / 256, 256, 0, stream>>>(woh, whT + 2 * whg, Hh);
  k_cvt_w<<<(Hh * Hh) / 256, 256, 0, stream>>>(wch, whT + 3 * whg, Hh);

  // 4) persistent fused LSTM over all T steps
  k_lstm<<<NWG, 256, 0, stream>>>(xbf, whT, wxT, bf, bi, bo, bc, hbuf, bar, out);
}
